// CPFactorizedVoxelGrid_69423851373082
// MI455X (gfx1250) — matrix operand order repaired and checked by reading, not compiled, not run on hardware
//
#include <hip/hip_runtime.h>
#include <cstdint>

typedef __attribute__((ext_vector_type(16))) __bf16 v16bf;
typedef __attribute__((ext_vector_type(8)))  __bf16 v8bf;   // 16 bytes
typedef __attribute__((ext_vector_type(8)))  float  v8f;

#define C_DIM 64
#define F_DIM 32
#define L_DIM 512
#define BLOCK 256
#define PTS_PER_PASS 128                 // 8 waves * 16 points
#define ITERS 8
#define PTS_PER_BLOCK (PTS_PER_PASS * ITERS)

union AOp {
  v16bf v;
  v8bf  h[2];
  __bf16 e[16];
};

struct Axis {
  const __bf16* s;
  int r0, r1;
  __bf16 w0, w1;
};

static __device__ __forceinline__ Axis make_axis(const __bf16* s, float pt) {
  float xf = (pt + 1.0f) * 0.5f * (float)(L_DIM - 1);
  float x0 = floorf(xf);
  float w1 = xf - x0;
  int i0 = (int)x0;
  int i1 = i0 + 1;
  float w0 = 1.0f - w1;
  if (i0 < 0 || i0 >= L_DIM) w0 = 0.0f;   // zeros padding outside grid
  if (i1 < 0 || i1 >= L_DIM) w1 = 0.0f;
  int c0 = min(max(i0, 0), L_DIM - 1);
  int c1 = min(max(i1, 0), L_DIM - 1);
  Axis a;
  a.s = s;
  a.r0 = c0 * C_DIM;
  a.r1 = c1 * C_DIM;
  a.w0 = (__bf16)w0;
  a.w1 = (__bf16)w1;
  return a;
}

// Interpolate 8 consecutive channels [c0, c0+8) on all three axes and
// multiply — entirely in packed bf16 (V_PK_FMA_BF16 / V_PK_MUL_BF16).
// The result is already in packed A-operand element order.
static __device__ __forceinline__ v8bf interp8(const Axis& X, const Axis& Y,
                                               const Axis& Z, int c0) {
  v8bf x0 = *(const v8bf*)(X.s + X.r0 + c0);
  v8bf x1 = *(const v8bf*)(X.s + X.r1 + c0);
  v8bf y0 = *(const v8bf*)(Y.s + Y.r0 + c0);
  v8bf y1 = *(const v8bf*)(Y.s + Y.r1 + c0);
  v8bf z0 = *(const v8bf*)(Z.s + Z.r0 + c0);
  v8bf z1 = *(const v8bf*)(Z.s + Z.r1 + c0);
  v8bf fx = x0 * X.w0 + x1 * X.w1;
  v8bf fy = y0 * Y.w0 + y1 * Y.w1;
  v8bf fz = z0 * Z.w0 + z1 * Z.w1;
  return fx * fy * fz;
}

__global__ __launch_bounds__(BLOCK) void cp_voxel_wmma_kernel(
    const float* __restrict__ points,
    const float* __restrict__ vxg,
    const float* __restrict__ vyg,
    const float* __restrict__ vzg,
    const float* __restrict__ basis,
    float* __restrict__ out,
    int P) {
  // Line tables transposed to (L, C) in bf16: 3 * 512*64*2 = 192 KB LDS.
  __shared__ alignas(16) __bf16 sVx[L_DIM * C_DIM];
  __shared__ alignas(16) __bf16 sVy[L_DIM * C_DIM];
  __shared__ alignas(16) __bf16 sVz[L_DIM * C_DIM];

  for (int idx = threadIdx.x; idx < L_DIM * C_DIM; idx += BLOCK) {
    int c = idx >> 9;          // idx / L_DIM   (source is (C, L) row-major)
    int x = idx & (L_DIM - 1); // idx % L_DIM
    sVx[x * C_DIM + c] = (__bf16)vxg[idx];
    sVy[x * C_DIM + c] = (__bf16)vyg[idx];
    sVz[x * C_DIM + c] = (__bf16)vzg[idx];
  }
  __syncthreads();

  const int lane = threadIdx.x & 31;
  const int wave = threadIdx.x >> 5;
  const int m    = lane & 15;   // A-row / D-column within tile
  const int half = lane >> 4;   // 0: K-low half, 1: K-high half

  // Loop-invariant B operands: B[kc][nt] covers K = kc*32..+31, N = nt*16..+15.
  // 16-bit B 32x16 layout: lanes 0-15 hold K {0..7,16..23}, lanes 16-31 hold K {8..15,24..31},
  // VGPR i packs K = {half*8+2i+(i>=4?8:0), +1}; column N = lane%16.
  AOp B[2][2];
#pragma unroll
  for (int kc = 0; kc < 2; ++kc)
#pragma unroll
    for (int nt = 0; nt < 2; ++nt)
#pragma unroll
      for (int i = 0; i < 8; ++i) {
        int c = kc * 32 + half * 8 + 2 * i + (i >= 4 ? 8 : 0);
        int f = nt * 16 + m;
        B[kc][nt].e[2 * i]     = (__bf16)basis[c * F_DIM + f];
        B[kc][nt].e[2 * i + 1] = (__bf16)basis[(c + 1) * F_DIM + f];
      }

  for (int it = 0; it < ITERS; ++it) {
    int pbase = blockIdx.x * PTS_PER_BLOCK + it * PTS_PER_PASS + wave * 16;
    if (pbase >= P) break;

    int p  = pbase + m;
    int pc = p < P ? p : P - 1;
    float px = points[(size_t)pc * 3 + 0];
    float py = points[(size_t)pc * 3 + 1];
    float pz = points[(size_t)pc * 3 + 2];
    if (it + 1 < ITERS)
      __builtin_prefetch(&points[(size_t)(pc + PTS_PER_PASS) * 3], 0, 1);

    Axis X = make_axis(sVx, px);
    Axis Y = make_axis(sVy, py);
    Axis Z = make_axis(sVz, pz);

    // A fragments. 16-bit A 16x32 layout: lanes 0-15 hold row M=lane with K
    // runs {0..7} (V0-3) and {16..23} (V4-7); lanes 16-31 hold runs
    // {8..15},{24..31} — i.e. two contiguous 8-channel runs per K-chunk.
    const int o = half * 8;
    AOp A0, A1;
    A0.h[0] = interp8(X, Y, Z,  0 + o);
    A0.h[1] = interp8(X, Y, Z, 16 + o);
    A1.h[0] = interp8(X, Y, Z, 32 + o);
    A1.h[1] = interp8(X, Y, Z, 48 + o);

    v8f acc0 = {}, acc1 = {};
    acc0 = __builtin_amdgcn_wmma_f32_16x16x32_bf16(false, A0.v, false, B[0][0].v,
                                                   (short)0, acc0, false, false);
    acc0 = __builtin_amdgcn_wmma_f32_16x16x32_bf16(false, A1.v, false, B[1][0].v,
                                                   (short)0, acc0, false, false);
    acc1 = __builtin_amdgcn_wmma_f32_16x16x32_bf16(false, A0.v, false, B[0][1].v,
                                                   (short)0, acc1, false, false);
    acc1 = __builtin_amdgcn_wmma_f32_16x16x32_bf16(false, A1.v, false, B[1][1].v,
                                                   (short)0, acc1, false, false);

    // D layout: VGPR r, lane l -> M = r + 8*(l/16), N = l%16.
    if (pbase + 16 <= P) {
      // Full tile (the common case): one base address, immediate offsets.
      float* orow = out + (size_t)(pbase + half * 8) * F_DIM + m;
#pragma unroll
      for (int r = 0; r < 8; ++r) {
        orow[r * F_DIM]      = acc0[r];
        orow[r * F_DIM + 16] = acc1[r];
      }
    } else {
#pragma unroll
      for (int r = 0; r < 8; ++r) {
        long long pp = (long long)pbase + r + half * 8;
        if (pp < P) {
          out[(size_t)pp * F_DIM + m]      = acc0[r];
          out[(size_t)pp * F_DIM + 16 + m] = acc1[r];
        }
      }
    }
  }
}

extern "C" void kernel_launch(void* const* d_in, const int* in_sizes, int n_in,
                              void* d_out, int out_size, void* d_ws, size_t ws_size,
                              hipStream_t stream) {
  const float* points = (const float*)d_in[0];
  const float* vx     = (const float*)d_in[1];
  const float* vy     = (const float*)d_in[2];
  const float* vz     = (const float*)d_in[3];
  const float* basis  = (const float*)d_in[4];
  float* out = (float*)d_out;

  int P = in_sizes[0] / 3;  // points is (G=1, P, 3)
  int grid = (P + PTS_PER_BLOCK - 1) / PTS_PER_BLOCK;
  if (grid < 1) grid = 1;

  cp_voxel_wmma_kernel<<<grid, BLOCK, 0, stream>>>(points, vx, vy, vz, basis,
                                                   out, P);
}
